// QCNet_37761352466458
// MI455X (gfx1250) — matrix operand order repaired and checked by reading, not compiled, not run on hardware
//
#include <hip/hip_runtime.h>
#include <hip/hip_bf16.h>

// ---- problem constants (from reference) ----
#define B_SZ 256
#define D_SZ 25200        // C*H*W
#define HID  512
#define KP   25216        // D_SZ padded up to multiple of 32 for K of GEMM1
#define TOPK 5040         // int(0.2 * 100 * 252)
#define MBLK 4            // 4 m-tiles (64 rows) per wave, reusing the B fragment

typedef __attribute__((ext_vector_type(16))) __bf16 v16bf;
typedef __attribute__((ext_vector_type(8)))  float  v8f;

// =========================================================================
// Fragment layouts (wave32, ISA 7.12.2):
//  A-frag flat index: (((m_tile*kblocks + kb)*32 + lane)*16 + e)
//    m = m_tile*16 + (lane & 15)
//    k = kb*32 + ((e >> 3) << 4) + ((lane >> 4) << 3) + (e & 7)
//  B-frag flat index: (((n_tile*kblocks + kb)*32 + lane)*16 + e)
//    n = n_tile*16 + (lane & 15)
//    k = kb*32 + ((e >> 3) << 4) + ((lane >> 4) << 3) + (e & 7)
// Both give each lane one contiguous 32-byte chunk per fragment -> all K-loop
// loads are base + immediate offset (deep compiler pipelining, s_clause pairs).
// =========================================================================

// -------------------------------------------------------------------------
// K0: x = com*mask + pair -> bf16, written directly in A-fragment layout,
// K-padded D_SZ -> KP with zeros.
// -------------------------------------------------------------------------
__global__ void fuse_pack_x(const float* __restrict__ com,
                            const float* __restrict__ pair,
                            const float* __restrict__ mask,
                            __bf16* __restrict__ dst) {
  int idx = blockIdx.x * blockDim.x + threadIdx.x;   // over B_SZ*KP
  const int kblocks = KP >> 5;
  int e = idx & 15;
  int lane = (idx >> 4) & 31;
  int t = idx >> 9;
  int kb = t % kblocks;
  int m_tile = t / kblocks;
  if (m_tile >= (B_SZ >> 4)) return;
  int m = m_tile * 16 + (lane & 15);
  int k = kb * 32 + ((e >> 3) << 4) + ((lane >> 4) << 3) + (e & 7);
  float v = 0.0f;
  if (k < D_SZ) {
    int g = m * D_SZ + k;
    v = com[g] * mask[g] + pair[g];
  }
  dst[idx] = (__bf16)v;
}

// -------------------------------------------------------------------------
// K1: repack a row-major bf16 activation [M x K] into A-fragment layout.
// (tiny: 256x512 elements)
// -------------------------------------------------------------------------
__global__ void pack_a_bf16(const __bf16* __restrict__ src,
                            __bf16* __restrict__ dst, int M, int K) {
  int idx = blockIdx.x * blockDim.x + threadIdx.x;
  if (idx >= M * K) return;
  const int kblocks = K >> 5;
  int e = idx & 15;
  int lane = (idx >> 4) & 31;
  int t = idx >> 9;
  int kb = t % kblocks;
  int m_tile = t / kblocks;
  int m = m_tile * 16 + (lane & 15);
  int k = kb * 32 + ((e >> 3) << 4) + ((lane >> 4) << 3) + (e & 7);
  dst[idx] = src[m * K + k];
}

// -------------------------------------------------------------------------
// K2: pack W [Kreal x N] (f32, row-major) into bf16 B-fragment layout.
// Rows k >= Kreal are zero-filled (K padding).
// -------------------------------------------------------------------------
__global__ void pack_w_bf16(const float* __restrict__ W,
                            __bf16* __restrict__ dst,
                            int Kreal, int KpK, int N) {
  long long idx = (long long)blockIdx.x * blockDim.x + threadIdx.x;
  long long total = (long long)KpK * N;
  if (idx >= total) return;
  int e = (int)(idx & 15);
  long long t = idx >> 4;
  int lane = (int)(t & 31);
  t >>= 5;
  int kblocks = KpK >> 5;
  int kb = (int)(t % kblocks);
  int n_tile = (int)(t / kblocks);
  int k = kb * 32 + ((e >> 3) << 4) + ((lane >> 4) << 3) + (e & 7);
  int n = n_tile * 16 + (lane & 15);
  float v = (k < Kreal) ? W[(long long)k * N + n] : 0.0f;
  dst[idx] = (__bf16)v;
}

// -------------------------------------------------------------------------
// K3: GEMM (packed-A bf16 x packed-B bf16) + bias + ReLU, bf16 row-major out.
// One wave computes a 64x16 strip (MBLK=4 accumulators); B fragment loaded
// once per K-step and reused 4x. All loop loads are base + immediate.
// Wave-uniform early-exit keeps EXEC all-ones for WMMA.
// -------------------------------------------------------------------------
__global__ void gemm_wmma_bias_relu_bf16out(const __bf16* __restrict__ Ap,
                                            const __bf16* __restrict__ Bp,
                                            const float* __restrict__ bias,
                                            __bf16* __restrict__ C,
                                            int M, int N, int Kp_) {
  int lane = threadIdx.x & 31;
  int wid = (int)((blockIdx.x * (long long)blockDim.x + threadIdx.x) >> 5);
  int tiles_n = N >> 4;
  int strips_m = M >> 6;
  int strips = strips_m * tiles_n;
  if (wid >= strips) return;                // wave-uniform
  int strip_m = wid / tiles_n;
  int tile_n = wid - strip_m * tiles_n;
  int kblocks = Kp_ >> 5;
  const long long astep = (long long)kblocks * 512;   // elems per m_tile
  const __bf16* bptr = Bp + ((long long)tile_n * kblocks * 32 + lane) * 16;
  const __bf16* aptr = Ap + ((long long)(strip_m * MBLK) * kblocks * 32 + lane) * 16;
  v8f acc[MBLK] = {};
  for (int kb = 0; kb < kblocks; ++kb) {
    v16bf b = *(const v16bf*)(bptr + (long long)kb * 512);
    __builtin_prefetch(bptr + (long long)(kb + 2) * 512, 0, 1);
#pragma unroll
    for (int mi = 0; mi < MBLK; ++mi) {
      v16bf a = *(const v16bf*)(aptr + astep * mi + (long long)kb * 512);
      acc[mi] = __builtin_amdgcn_wmma_f32_16x16x32_bf16(
          false, a, false, b, (short)0, acc[mi], false, false);
    }
  }
  int n = tile_n * 16 + (lane & 15);
  float bv = bias[n];
  int mbase = strip_m * 64 + ((lane >> 4) << 3);
#pragma unroll
  for (int mi = 0; mi < MBLK; ++mi) {
#pragma unroll
    for (int v = 0; v < 8; ++v) {
      float t = acc[mi][v] + bv;
      t = t > 0.0f ? t : 0.0f;
      C[(long long)(mbase + mi * 16 + v) * N + n] = (__bf16)t;
    }
  }
}

// -------------------------------------------------------------------------
// K4: GEMM + bias + sigmoid, f32 row-major output (qcm). Same 64x16 strip.
// -------------------------------------------------------------------------
__global__ void gemm_wmma_bias_sigmoid_f32out(const __bf16* __restrict__ Ap,
                                              const __bf16* __restrict__ Bp,
                                              const float* __restrict__ bias,
                                              float* __restrict__ C,
                                              int M, int N, int Kp_) {
  int lane = threadIdx.x & 31;
  int wid = (int)((blockIdx.x * (long long)blockDim.x + threadIdx.x) >> 5);
  int tiles_n = N >> 4;
  int strips_m = M >> 6;
  int strips = strips_m * tiles_n;
  if (wid >= strips) return;                // wave-uniform
  int strip_m = wid / tiles_n;
  int tile_n = wid - strip_m * tiles_n;
  int kblocks = Kp_ >> 5;
  const long long astep = (long long)kblocks * 512;
  const __bf16* bptr = Bp + ((long long)tile_n * kblocks * 32 + lane) * 16;
  const __bf16* aptr = Ap + ((long long)(strip_m * MBLK) * kblocks * 32 + lane) * 16;
  v8f acc[MBLK] = {};
  for (int kb = 0; kb < kblocks; ++kb) {
    v16bf b = *(const v16bf*)(bptr + (long long)kb * 512);
    __builtin_prefetch(bptr + (long long)(kb + 2) * 512, 0, 1);
#pragma unroll
    for (int mi = 0; mi < MBLK; ++mi) {
      v16bf a = *(const v16bf*)(aptr + astep * mi + (long long)kb * 512);
      acc[mi] = __builtin_amdgcn_wmma_f32_16x16x32_bf16(
          false, a, false, b, (short)0, acc[mi], false, false);
    }
  }
  int n = tile_n * 16 + (lane & 15);
  float bv = bias[n];
  int mbase = strip_m * 64 + ((lane >> 4) << 3);
#pragma unroll
  for (int mi = 0; mi < MBLK; ++mi) {
#pragma unroll
    for (int v = 0; v < 8; ++v) {
      float t = acc[mi][v] + bv;
      C[(long long)(mbase + mi * 16 + v) * N + n] = 1.0f / (1.0f + __expf(-t));
    }
  }
}

// -------------------------------------------------------------------------
// K5: deterministic per-row top-k -> binary mask (LDS histogram + ordered
// ranking in the threshold bin).
// -------------------------------------------------------------------------
__device__ __forceinline__ int qbin(float v) {
  int b = (int)(v * 1024.0f);
  return b < 0 ? 0 : (b > 1023 ? 1023 : b);
}

__global__ void topk_mask_kernel(const float* __restrict__ qcm,
                                 float* __restrict__ out, int Dtot, int k) {
  __shared__ int hist[1024];
  __shared__ int cnts[256];
  __shared__ int s_thresh, s_need;
  int row = blockIdx.x;
  const float* q = qcm + (long long)row * Dtot;
  float* o = out + (long long)row * Dtot;

  for (int i = threadIdx.x; i < 1024; i += blockDim.x) hist[i] = 0;
  __syncthreads();
  for (int i = threadIdx.x; i < Dtot; i += blockDim.x)
    atomicAdd(&hist[qbin(q[i])], 1);
  __syncthreads();

  if (threadIdx.x == 0) {
    int acc = 0, t = 0;
    for (int b = 1023; b >= 0; --b) {
      t = b;
      if (acc + hist[b] >= k) break;
      acc += hist[b];
    }
    s_thresh = t;
    s_need = k - acc;   // how many to take from the threshold bin
  }
  __syncthreads();
  int thresh = s_thresh, need = s_need;

  int my = 0;
  for (int i = threadIdx.x; i < Dtot; i += blockDim.x)
    if (qbin(q[i]) == thresh) my++;
  cnts[threadIdx.x] = my;
  __syncthreads();
  if (threadIdx.x == 0) {
    int acc = 0;
    for (int t2 = 0; t2 < 256; ++t2) { int c = cnts[t2]; cnts[t2] = acc; acc += c; }
  }
  __syncthreads();

  int rank = cnts[threadIdx.x];
  for (int i = threadIdx.x; i < Dtot; i += blockDim.x) {
    int b = qbin(q[i]);
    float r;
    if (b > thresh) r = 1.0f;
    else if (b == thresh) { r = (rank < need) ? 1.0f : 0.0f; rank++; }
    else r = 0.0f;
    o[i] = r;
  }
}

// -------------------------------------------------------------------------
// launch
// -------------------------------------------------------------------------
extern "C" void kernel_launch(void* const* d_in, const int* in_sizes, int n_in,
                              void* d_out, int out_size, void* d_ws, size_t ws_size,
                              hipStream_t stream) {
  (void)in_sizes; (void)n_in; (void)out_size; (void)ws_size;
  const float* com  = (const float*)d_in[0];
  const float* pair = (const float*)d_in[1];
  const float* mask = (const float*)d_in[2];
  const float* W1 = (const float*)d_in[3];
  const float* b1 = (const float*)d_in[4];
  const float* W2 = (const float*)d_in[5];
  const float* b2 = (const float*)d_in[6];
  const float* W3 = (const float*)d_in[7];
  const float* b3 = (const float*)d_in[8];
  float* out = (float*)d_out;

  // workspace carve-up (~92 MB total)
  char* ws = (char*)d_ws;
  size_t off = 0;
  auto carve = [&](size_t bytes) -> void* {
    void* p = ws + off;
    off += (bytes + 255) & ~(size_t)255;
    return p;
  };
  __bf16* xpk  = (__bf16*)carve((size_t)B_SZ * KP * 2);     // packed A (GEMM1)
  __bf16* w1p  = (__bf16*)carve((size_t)KP * HID * 2);
  __bf16* w2p  = (__bf16*)carve((size_t)HID * HID * 2);
  __bf16* w3p  = (__bf16*)carve((size_t)HID * D_SZ * 2);
  __bf16* x1bf = (__bf16*)carve((size_t)B_SZ * HID * 2);    // GEMM1 out (row-major)
  __bf16* x1pk = (__bf16*)carve((size_t)B_SZ * HID * 2);    // packed A (GEMM2)
  __bf16* x2bf = (__bf16*)carve((size_t)B_SZ * HID * 2);    // GEMM2 out (row-major)
  __bf16* x2pk = (__bf16*)carve((size_t)B_SZ * HID * 2);    // packed A (GEMM3)
  float*  qcm  = (float*)carve((size_t)B_SZ * D_SZ * 4);

  const int thr = 256;  // 8 waves/block on wave32

  fuse_pack_x<<<(B_SZ * KP) / thr, thr, 0, stream>>>(com, pair, mask, xpk);

  pack_w_bf16<<<(int)(((size_t)KP * HID) / thr), thr, 0, stream>>>(W1, w1p, D_SZ, KP, HID);
  pack_w_bf16<<<(HID * HID) / thr, thr, 0, stream>>>(W2, w2p, HID, HID, HID);
  pack_w_bf16<<<(int)(((size_t)HID * D_SZ) / thr), thr, 0, stream>>>(W3, w3p, HID, HID, D_SZ);

  // GEMM1: relu(x @ W1 + b1) -> x1
  int waves12 = (B_SZ / 64) * (HID / 16);         // 128 waves (64x16 strips)
  int blocks12 = (waves12 * 32 + thr - 1) / thr;
  gemm_wmma_bias_relu_bf16out<<<blocks12, thr, 0, stream>>>(
      xpk, w1p, b1, x1bf, B_SZ, HID, KP);
  pack_a_bf16<<<(B_SZ * HID) / thr, thr, 0, stream>>>(x1bf, x1pk, B_SZ, HID);

  // GEMM2: relu(x1 @ W2 + b2) -> x2
  gemm_wmma_bias_relu_bf16out<<<blocks12, thr, 0, stream>>>(
      x1pk, w2p, b2, x2bf, B_SZ, HID, HID);
  pack_a_bf16<<<(B_SZ * HID) / thr, thr, 0, stream>>>(x2bf, x2pk, B_SZ, HID);

  // GEMM3: sigmoid(x2 @ W3 + b3) -> qcm (f32)
  int waves3 = (B_SZ / 64) * (D_SZ / 16);         // 6300 waves
  int blocks3 = (waves3 * 32 + thr - 1) / thr;
  gemm_wmma_bias_sigmoid_f32out<<<blocks3, thr, 0, stream>>>(
      x2pk, w3p, b3, qcm, B_SZ, D_SZ, HID);

  // top-k binary mask per row
  topk_mask_kernel<<<B_SZ, 256, 0, stream>>>(qcm, out, D_SZ, TOPK);
}
